// GFootballMamba_38173669327500
// MI455X (gfx1250) — matrix operand order repaired and checked by reading, not compiled, not run on hardware
//
#include <hip/hip_runtime.h>

// ---------------------------------------------------------------------------
// GFootball Mamba forward on MI455X (gfx1250): f16 WMMA GEMMs + f32 scan
// ---------------------------------------------------------------------------

typedef _Float16 half8 __attribute__((ext_vector_type(8)));
typedef _Float16 v16h  __attribute__((ext_vector_type(16)));
typedef float    v8f   __attribute__((ext_vector_type(8)));

#define MTOK   (32 * 512)    // B*L tokens
#define DMODEL 512

__device__ __forceinline__ v16h cat16(half8 lo, half8 hi)
{
    return __builtin_shufflevector(lo, hi,
        0,1,2,3,4,5,6,7,8,9,10,11,12,13,14,15);
}

// ---------------------------------------------------------------------------
// Repack f32 [K,N] row-major weights into f16 WMMA B-fragment order:
//   tile (kt, nt): 32 lanes x 16 halves, lane L: n = nt*16 + L%16,
//   halves i: k = kt*32 + (L/16)*16 + i.  Zero pad outside [K,N].
// ---------------------------------------------------------------------------
__global__ void repack_w(const float* __restrict__ W, int K, int N, int ldw,
                         _Float16* __restrict__ out, int K_tiles, int N_tiles)
{
    size_t idx = (size_t)blockIdx.x * blockDim.x + threadIdx.x;
    size_t total = (size_t)K_tiles * N_tiles * 512;
    if (idx >= total) return;
    int i    = (int)(idx & 15);
    int lane = (int)((idx >> 4) & 31);
    int tile = (int)(idx >> 9);
    int nt = tile % N_tiles;
    int kt = tile / N_tiles;
    int k = kt * 32 + (lane >> 4) * 16 + i;
    int n = nt * 16 + (lane & 15);
    float v = (k < K && n < N) ? W[(size_t)k * ldw + n] : 0.0f;
    out[idx] = (_Float16)v;
}

// ---------------------------------------------------------------------------
// Main GEMM: C[M,N] = A[M,K] * B[K,N] (+bias)(+resid), f16 in, f32 acc.
// Block = 256 threads (8 wave32), block tile 256x64. Each wave owns a
// 32x64 strip: acc[2][4], so every B fragment feeds 2 WMMAs.
// K loop: ping-pong register double-buffering, unrolled by 2 — no
// fragment rotation moves, and in-order load returns let the waitcnt
// pass overlap one tile's loads with the other tile's WMMA burst.
// ---------------------------------------------------------------------------
__global__ __launch_bounds__(256) void gemm_wmma(
    const _Float16* __restrict__ A, int lda,
    const _Float16* __restrict__ Bp, int n_tiles_total,
    const float* __restrict__ bias,
    const float* __restrict__ resid, int ldr,
    float* __restrict__ C, int ldc, int col_off, int n_valid,
    _Float16* __restrict__ C16, int ldc16,
    int K_tiles)
{
    const int wave = threadIdx.x >> 5;
    const int lane = threadIdx.x & 31;
    const int hi   = lane >> 4;            // which K-half this lane owns
    const int m0   = blockIdx.x * 256 + wave * 32;
    const int row0 = m0 + (lane & 15);
    const int nt0  = blockIdx.y * 4;

    v8f acc[2][4] = {};

    // A fragment: lane<16 -> K {k0..k0+15}
    //             lane>=16 -> K {k0+16..k0+31}
    const _Float16* arow0 = A + (size_t)row0 * lda + hi * 16;
    const _Float16* arow1 = arow0 + (size_t)16 * lda;
    const _Float16* bbase = Bp + (size_t)lane * 16;

    auto loadA = [&](const _Float16* p) -> v16h {
        half8 lo  = *(const half8*)(p);
        half8 hi8 = *(const half8*)(p + 8);
        return cat16(lo, hi8);
    };
    auto loadB = [&](int kt, int j) -> v16h {
        const _Float16* p = bbase + (size_t)(kt * n_tiles_total + nt0 + j) * 512;
        half8 lo  = *(const half8*)(p);
        half8 hi8 = *(const half8*)(p + 8);
        return cat16(lo, hi8);
    };

#define LOAD_TILE(KT, a0, a1, b0, b1, b2, b3)              \
    {                                                      \
        const int _k = (KT) * 32;                          \
        a0 = loadA(arow0 + _k);                            \
        a1 = loadA(arow1 + _k);                            \
        b0 = loadB((KT), 0);                               \
        b1 = loadB((KT), 1);                               \
        b2 = loadB((KT), 2);                               \
        b3 = loadB((KT), 3);                               \
    }

#define COMPUTE(a0, a1, b0, b1, b2, b3)                                       \
    {                                                                         \
        acc[0][0] = __builtin_amdgcn_wmma_f32_16x16x32_f16(                   \
            false, a0, false, b0, (short)0, acc[0][0], false, false);         \
        acc[1][0] = __builtin_amdgcn_wmma_f32_16x16x32_f16(                   \
            false, a1, false, b0, (short)0, acc[1][0], false, false);         \
        acc[0][1] = __builtin_amdgcn_wmma_f32_16x16x32_f16(                   \
            false, a0, false, b1, (short)0, acc[0][1], false, false);         \
        acc[1][1] = __builtin_amdgcn_wmma_f32_16x16x32_f16(                   \
            false, a1, false, b1, (short)0, acc[1][1], false, false);         \
        acc[0][2] = __builtin_amdgcn_wmma_f32_16x16x32_f16(                   \
            false, a0, false, b2, (short)0, acc[0][2], false, false);         \
        acc[1][2] = __builtin_amdgcn_wmma_f32_16x16x32_f16(                   \
            false, a1, false, b2, (short)0, acc[1][2], false, false);         \
        acc[0][3] = __builtin_amdgcn_wmma_f32_16x16x32_f16(                   \
            false, a0, false, b3, (short)0, acc[0][3], false, false);         \
        acc[1][3] = __builtin_amdgcn_wmma_f32_16x16x32_f16(                   \
            false, a1, false, b3, (short)0, acc[1][3], false, false);         \
    }

    // Ping-pong fragment sets (fixed registers, no rotation moves).
    v16h pa0, pa1, pb0, pb1, pb2, pb3;   // even tiles
    v16h qa0, qa1, qb0, qb1, qb2, qb3;   // odd tiles

    LOAD_TILE(0, pa0, pa1, pb0, pb1, pb2, pb3);
    int kt = 0;
    for (; kt + 2 <= K_tiles; kt += 2) {
        LOAD_TILE(kt + 1, qa0, qa1, qb0, qb1, qb2, qb3);
        COMPUTE(pa0, pa1, pb0, pb1, pb2, pb3);
        if (kt + 2 < K_tiles)
            LOAD_TILE(kt + 2, pa0, pa1, pb0, pb1, pb2, pb3);
        COMPUTE(qa0, qa1, qb0, qb1, qb2, qb3);
    }
    if (kt < K_tiles)                    // odd K_tiles tail (tile kt preloaded)
        COMPUTE(pa0, pa1, pb0, pb1, pb2, pb3);

#undef LOAD_TILE
#undef COMPUTE

    // Epilogue. C/D layout: n_local = lane&15, m_local = r + 8*(lane>=16)
#pragma unroll
    for (int rb = 0; rb < 2; ++rb) {
#pragma unroll
        for (int j = 0; j < 4; ++j) {
            const int ncol = (nt0 + j) * 16 + (lane & 15);
            if (ncol >= n_valid) continue;
            const float bv = bias ? bias[ncol] : 0.0f;
#pragma unroll
            for (int r = 0; r < 8; ++r) {
                const int m = m0 + rb * 16 + r + hi * 8;
                float v = acc[rb][j][r] + bv;
                if (resid) v += resid[(size_t)m * ldr + col_off + ncol];
                C[(size_t)m * ldc + col_off + ncol] = v;
                if (C16) C16[(size_t)m * ldc16 + col_off + ncol] = (_Float16)v;
            }
        }
    }
}

// ---------------------------------------------------------------------------
// Row LayerNorm (+optional SiLU). One block (256 thr) per row.
// ---------------------------------------------------------------------------
__global__ __launch_bounds__(256) void ln_act(
    const float* __restrict__ X, int ldx,
    const float* __restrict__ w, const float* __restrict__ b,
    int N, int do_silu,
    float* __restrict__ out32, int ldo32,
    _Float16* __restrict__ out16, int ldo16, int off16)
{
    const int row = blockIdx.x;
    const float* xr = X + (size_t)row * ldx;
    float s = 0.f, ss = 0.f;
    for (int i = threadIdx.x; i < N; i += 256) {
        float v = xr[i]; s += v; ss += v * v;
    }
#pragma unroll
    for (int o = 16; o > 0; o >>= 1) {
        s  += __shfl_down(s,  o, 32);
        ss += __shfl_down(ss, o, 32);
    }
    __shared__ float red[2][8];
    const int wv = threadIdx.x >> 5, ln = threadIdx.x & 31;
    if (ln == 0) { red[0][wv] = s; red[1][wv] = ss; }
    __syncthreads();
    if (threadIdx.x == 0) {
        float a = 0.f, c = 0.f;
#pragma unroll
        for (int i = 0; i < 8; ++i) { a += red[0][i]; c += red[1][i]; }
        red[0][0] = a; red[1][0] = c;
    }
    __syncthreads();
    const float mean = red[0][0] / N;
    const float var  = red[1][0] / N - mean * mean;
    const float rstd = rsqrtf(var + 1e-5f);
    for (int i = threadIdx.x; i < N; i += 256) {
        float v = (xr[i] - mean) * rstd * w[i] + b[i];
        if (do_silu) v = v / (1.f + expf(-v));
        if (out32) out32[(size_t)row * ldo32 + i] = v;
        if (out16) out16[(size_t)row * ldo16 + off16 + i] = (_Float16)v;
    }
}

// ---------------------------------------------------------------------------
// Front-end prep: f32 obs -> f16 padded [M,128]; embed gather into concat.
// ---------------------------------------------------------------------------
__global__ void prep_obs(const float* __restrict__ obs, _Float16* __restrict__ o16)
{
    size_t idx = (size_t)blockIdx.x * blockDim.x + threadIdx.x;
    if (idx >= (size_t)MTOK * 128) return;
    int col = (int)(idx & 127);
    size_t tok = idx >> 7;
    float v = (col < 115) ? obs[tok * 115 + col] : 0.0f;
    o16[idx] = (_Float16)v;
}

__global__ void prep_emb(const int* __restrict__ acts,
                         const float* __restrict__ embed,
                         _Float16* __restrict__ cat)   // [M,544], cols 512..543
{
    size_t idx = (size_t)blockIdx.x * blockDim.x + threadIdx.x;
    if (idx >= (size_t)MTOK * 32) return;
    int e = (int)(idx & 31);
    size_t tok = idx >> 5;
    float v = (e < 16) ? embed[(size_t)acts[tok] * 16 + e] : 0.0f;
    cat[tok * 544 + 512 + e] = (_Float16)v;
}

// ---------------------------------------------------------------------------
// S6 selective scan. Thread owns one (b,d) pair, h[16] in registers.
// Bt/Ct (32 floats per step) double-buffered in LDS, one barrier/step.
// ---------------------------------------------------------------------------
__global__ __launch_bounds__(256) void s6_scan(
    const float* __restrict__ gate,   // [M,1024], x_in = cols 0..511
    const float* __restrict__ dtraw,  // [M,512]  (dt_w proj + dt_b, pre-softplus)
    const float* __restrict__ BC,     // [M,32]   Bt cols 0..15, Ct cols 16..31
    const float* __restrict__ A_log,  // [512,16]
    float* __restrict__ Y)            // [M,512]
{
    const int b = blockIdx.x >> 1;
    const int d = (blockIdx.x & 1) * 256 + threadIdx.x;
    float A[16];
#pragma unroll
    for (int n = 0; n < 16; ++n) A[n] = -expf(A_log[(size_t)d * 16 + n]);
    float h[16] = {};
    __shared__ float bc[2][32];
    const size_t base = (size_t)b * 512;
    for (int l = 0; l < 512; ++l) {
        const size_t r = base + l;
        const int p = l & 1;
        if (threadIdx.x < 32) bc[p][threadIdx.x] = BC[r * 32 + threadIdx.x];
        __syncthreads();
        const float dv = dtraw[r * 512 + d];
        const float dt = (dv > 20.f) ? dv : log1pf(expf(dv));   // softplus
        const float du = dt * gate[r * 1024 + d];
        float y = 0.f;
#pragma unroll
        for (int n = 0; n < 16; ++n) {
            h[n] = expf(dt * A[n]) * h[n] + du * bc[p][n];
            y += h[n] * bc[p][16 + n];
        }
        Y[r * 512 + d] = y;
    }
}

// ssm = y * silu(z) + x_in * D  ->  f16 for out-projection GEMM
__global__ void ssm_mix(const float* __restrict__ Y,
                        const float* __restrict__ gate,
                        const float* __restrict__ D,
                        _Float16* __restrict__ out)
{
    size_t idx = (size_t)blockIdx.x * blockDim.x + threadIdx.x;
    if (idx >= (size_t)MTOK * 512) return;
    int d = (int)(idx & 511);
    size_t r = idx >> 9;
    float z = gate[r * 1024 + 512 + d];
    float xin = gate[r * 1024 + d];
    float v = Y[idx] * (z / (1.f + expf(-z))) + xin * D[d];
    out[idx] = (_Float16)v;
}

// ---------------------------------------------------------------------------
// Host orchestration
// ---------------------------------------------------------------------------
extern "C" void kernel_launch(void* const* d_in, const int* in_sizes, int n_in,
                              void* d_out, int out_size, void* d_ws, size_t ws_size,
                              hipStream_t stream)
{
    (void)in_sizes; (void)n_in; (void)out_size; (void)ws_size;
    const int M = MTOK;

    // ---- input pointers (setup_inputs dict order, nested params flattened) ----
    const float* obs    = (const float*)d_in[0];
    const int*   acts   = (const int*)  d_in[1];
    const float* mlp_w0 = (const float*)d_in[2];
    const float* mlp_b0 = (const float*)d_in[3];
    const float* ln0_w  = (const float*)d_in[4];
    const float* ln0_b  = (const float*)d_in[5];
    const float* mlp_w1 = (const float*)d_in[6];
    const float* mlp_b1 = (const float*)d_in[7];
    const float* ln1_w  = (const float*)d_in[8];
    const float* ln1_b  = (const float*)d_in[9];
    const float* embedw = (const float*)d_in[10];
    const float* in_w   = (const float*)d_in[11];
    const float* in_b   = (const float*)d_in[12];
    const float* inln_w = (const float*)d_in[13];
    const float* inln_b = (const float*)d_in[14];
    const float* fln_w  = (const float*)d_in[15];
    const float* fln_b  = (const float*)d_in[16];
    const float* pol_w0 = (const float*)d_in[17];
    const float* pol_b0 = (const float*)d_in[18];
    const float* pln_w  = (const float*)d_in[19];
    const float* pln_b  = (const float*)d_in[20];
    const float* pol_w1 = (const float*)d_in[21];
    const float* pol_b1 = (const float*)d_in[22];
    const float* val_w0 = (const float*)d_in[23];
    const float* val_b0 = (const float*)d_in[24];
    const float* vln_w  = (const float*)d_in[25];
    const float* vln_b  = (const float*)d_in[26];
    const float* val_w1 = (const float*)d_in[27];
    const float* val_b1 = (const float*)d_in[28];
    // per-layer: ln_w ln_b in_gate_w dt_w dt_b A_log B_w C_w D out_w
    auto LP = [&](int layer, int slot) -> const float* {
        return (const float*)d_in[29 + layer * 10 + slot];
    };

    // ---- workspace bump allocator ----
    char* ws = (char*)d_ws;
    size_t cur = 0;
    auto alloc = [&](size_t bytes) -> void* {
        void* p = ws + cur;
        cur += (bytes + 255) & ~(size_t)255;
        return p;
    };

    auto repack = [&](const float* W, int K, int N, int kt, int nt) -> _Float16* {
        _Float16* out = (_Float16*)alloc((size_t)kt * nt * 512 * sizeof(_Float16));
        size_t total = (size_t)kt * nt * 512;
        repack_w<<<dim3((unsigned)((total + 255) / 256)), 256, 0, stream>>>(
            W, K, N, N, out, kt, nt);
        return out;
    };

    // ---- pack all weights (L2-resident afterwards) ----
    _Float16* Wp_mlp0 = repack(mlp_w0, 115, 1024, 4, 64);
    _Float16* Wp_mlp1 = repack(mlp_w1, 1024, 512, 32, 32);
    _Float16* Wp_in   = repack(in_w,   528,  512, 17, 32);
    _Float16* Wp_gate[4]; _Float16* Wp_dt[4]; _Float16* Wp_B[4];
    _Float16* Wp_C[4];    _Float16* Wp_out[4];
    for (int i = 0; i < 4; ++i) {
        Wp_gate[i] = repack(LP(i, 2), 512, 1024, 16, 64);
        Wp_dt[i]   = repack(LP(i, 3), 512,  512, 16, 32);
        Wp_B[i]    = repack(LP(i, 6), 512,   16, 16, 4);
        Wp_C[i]    = repack(LP(i, 7), 512,   16, 16, 4);
        Wp_out[i]  = repack(LP(i, 9), 512,  512, 16, 32);
    }
    _Float16* Wp_p0 = repack(pol_w0, 512, 512, 16, 32);
    _Float16* Wp_p1 = repack(pol_w1, 512,  19, 16, 4);
    _Float16* Wp_v0 = repack(val_w0, 512, 512, 16, 32);
    _Float16* Wp_v1 = repack(val_w1, 512,  51, 16, 4);

    // ---- activation buffers ----
    _Float16* obs16 = (_Float16*)alloc((size_t)M * 128 * 2);
    _Float16* catb  = (_Float16*)alloc((size_t)M * 544 * 2);
    float*    t0    = (float*)   alloc((size_t)M * 1024 * 4); // wide f32 scratch / gate
    float*    t1    = (float*)   alloc((size_t)M * 512 * 4);  // f32 scratch / dtraw
    float*    x     = (float*)   alloc((size_t)M * 512 * 4);  // residual stream
    _Float16* xn16  = (_Float16*)alloc((size_t)M * 512 * 2);  // LN(x) / ssm f16
    _Float16* g16   = (_Float16*)alloc((size_t)M * 1024 * 2); // gate f16 / head f16
    float*    BC    = (float*)   alloc((size_t)M * 32 * 4);
    float*    Yb    = (float*)   alloc((size_t)M * 512 * 4);

    auto gemm = [&](const _Float16* A, int lda, const _Float16* Bp, int nt,
                    const float* bias, const float* resid, int ldr,
                    float* C, int ldc, int coloff, int nvalid,
                    _Float16* C16, int ldc16, int kt) {
        dim3 g(M / 256, nt / 4);
        gemm_wmma<<<g, 256, 0, stream>>>(A, lda, Bp, nt, bias, resid, ldr,
                                         C, ldc, coloff, nvalid, C16, ldc16, kt);
    };
    auto lnk = [&](const float* X, int ldx, const float* w, const float* b,
                   int N, int silu, float* o32, int ldo32,
                   _Float16* o16, int ldo16, int off16) {
        ln_act<<<M, 256, 0, stream>>>(X, ldx, w, b, N, silu,
                                      o32, ldo32, o16, ldo16, off16);
    };

    // ---- front-end MLP ----
    prep_obs<<<(M * 128 + 255) / 256, 256, 0, stream>>>(obs, obs16);
    gemm(obs16, 128, Wp_mlp0, 64, mlp_b0, nullptr, 0,
         t0, 1024, 0, 1024, nullptr, 0, 4);
    lnk(t0, 1024, ln0_w, ln0_b, 1024, 1, nullptr, 0, g16, 1024, 0);   // silu
    gemm(g16, 1024, Wp_mlp1, 32, mlp_b1, nullptr, 0,
         t1, 512, 0, 512, nullptr, 0, 32);
    lnk(t1, 512, ln1_w, ln1_b, 512, 0, nullptr, 0, catb, 544, 0);
    prep_emb<<<(M * 32 + 255) / 256, 256, 0, stream>>>(acts, embedw, catb);
    gemm(catb, 544, Wp_in, 32, in_b, nullptr, 0,
         t1, 512, 0, 512, nullptr, 0, 17);
    lnk(t1, 512, inln_w, inln_b, 512, 1, x, 512, nullptr, 0, 0);      // x = silu(ln(.))

    // ---- 4 S6 layers ----
    for (int i = 0; i < 4; ++i) {
        lnk(x, 512, LP(i, 0), LP(i, 1), 512, 0, nullptr, 0, xn16, 512, 0);
        gemm(xn16, 512, Wp_gate[i], 64, nullptr, nullptr, 0,
             t0, 1024, 0, 1024, g16, 1024, 16);                       // gate f32+f16
        gemm(g16, 1024, Wp_dt[i], 32, LP(i, 4), nullptr, 0,
             t1, 512, 0, 512, nullptr, 0, 16);                        // dtraw (pre-softplus)
        gemm(g16, 1024, Wp_B[i], 4, nullptr, nullptr, 0,
             BC, 32, 0, 16, nullptr, 0, 16);                          // Bt
        gemm(g16, 1024, Wp_C[i], 4, nullptr, nullptr, 0,
             BC, 32, 16, 16, nullptr, 0, 16);                         // Ct
        s6_scan<<<64, 256, 0, stream>>>(t0, t1, BC, LP(i, 5), Yb);
        ssm_mix<<<(M * 512 + 255) / 256, 256, 0, stream>>>(Yb, t0, LP(i, 8), xn16);
        gemm(xn16, 512, Wp_out[i], 32, nullptr, x, 512,
             x, 512, 0, 512, nullptr, 0, 16);                         // x += ssm @ out_w
    }

    // ---- heads ----
    float* out = (float*)d_out;                                       // [M,70]
    lnk(x, 512, fln_w, fln_b, 512, 0, nullptr, 0, xn16, 512, 0);
    gemm(xn16, 512, Wp_p0, 32, pol_b0, nullptr, 0,
         t1, 512, 0, 512, nullptr, 0, 16);
    lnk(t1, 512, pln_w, pln_b, 512, 1, nullptr, 0, g16, 512, 0);
    gemm(g16, 512, Wp_p1, 4, pol_b1, nullptr, 0,
         out, 70, 0, 19, nullptr, 0, 16);                             // policy
    gemm(xn16, 512, Wp_v0, 32, val_b0, nullptr, 0,
         t1, 512, 0, 512, nullptr, 0, 16);
    lnk(t1, 512, vln_w, vln_b, 512, 1, nullptr, 0, g16, 512, 0);
    gemm(g16, 512, Wp_v1, 4, val_b1, nullptr, 0,
         out, 70, 19, 51, nullptr, 0, 16);                            // value
}
